// DistanceEdgePredictor_69260642615903
// MI455X (gfx1250) — matrix operand order repaired and checked by reading, not compiled, not run on hardware
//
#include <hip/hip_runtime.h>
#include <hip/hip_bf16.h>

#define N_NODES 8192
#define NODE_DIM 64
#define HID 128
#define NT (N_NODES/16)                       // 512 tiles per dimension
#define M_PAIRS ((long long)N_NODES*(N_NODES-1)/2)   // 33,550,336

typedef __attribute__((ext_vector_type(16))) __bf16 v16bf;
typedef __attribute__((ext_vector_type(8)))  __bf16 v8bf;
typedef __attribute__((ext_vector_type(8)))  float  v8f;

__device__ __forceinline__ unsigned short f2bf_rne(float f) {
    unsigned int u = __float_as_uint(f);
    u += 0x7FFFu + ((u >> 16) & 1u);          // round-to-nearest-even
    return (unsigned short)(u >> 16);
}
__device__ __forceinline__ float bf2f(unsigned short s) {
    return __uint_as_float(((unsigned int)s) << 16);
}

// ---------------------------------------------------------------------------
// Kernel 1: h = relu(X @ W + b); emit bf16 hi/lo split of h and row norms sq.
// 8192x64 @ 64x128 = 134 MFLOP -- negligible; W staged once in LDS per block.
// ---------------------------------------------------------------------------
__global__ __launch_bounds__(128)
void encode_split_kernel(const float* __restrict__ X,
                         const float* __restrict__ W,
                         const float* __restrict__ b,
                         unsigned short* __restrict__ h_hi,
                         unsigned short* __restrict__ h_lo,
                         float* __restrict__ sq)
{
    __shared__ float Ws[NODE_DIM * HID];      // 32 KB
    __shared__ float xs[NODE_DIM];
    __shared__ float red[128];
    const int tid = threadIdx.x;              // = output column 0..127
    for (int i = tid; i < NODE_DIM * HID; i += 128) Ws[i] = W[i];
    const float bias = b[tid];
    const int row0 = blockIdx.x * 4;

    for (int r = 0; r < 4; ++r) {
        const int row = row0 + r;
        __syncthreads();                      // Ws ready / xs reusable
        if (tid < NODE_DIM) xs[tid] = X[(size_t)row * NODE_DIM + tid];
        __syncthreads();

        float acc = bias;
        #pragma unroll
        for (int k = 0; k < NODE_DIM; ++k)
            acc = fmaf(xs[k], Ws[k * HID + tid], acc);
        acc = fmaxf(acc, 0.0f);               // ReLU

        const unsigned short hi = f2bf_rne(acc);
        const float lo_f = acc - bf2f(hi);
        h_hi[(size_t)row * HID + tid] = hi;
        h_lo[(size_t)row * HID + tid] = f2bf_rne(lo_f);

        red[tid] = acc * acc;
        __syncthreads();
        for (int s = 64; s > 0; s >>= 1) {
            if (tid < s) red[tid] += red[tid + s];
            __syncthreads();
        }
        if (tid == 0) sq[row] = red[0];
    }
}

// ---------------------------------------------------------------------------
// Kernel 2: one wave32 per 16x16 tile of the NxN pairwise matrix.
// Gram via split-bf16 WMMA (hi*hi + hi*lo + lo*hi, f32 accumulate), then
// d2 = sq_i + sq_j - 2g -> sqrt -> sigmoid; stream soft_adj / probs / index.
// ---------------------------------------------------------------------------
__global__ __launch_bounds__(256)
void pair_wmma_kernel(const unsigned short* __restrict__ h_hi,
                      const unsigned short* __restrict__ h_lo,
                      const float* __restrict__ sq,
                      const float* __restrict__ thr_p,
                      float* __restrict__ out)
{
    const int lane = threadIdx.x;             // 0..31 (wave32)
    const int wave = threadIdx.y;             // 0..7
    const long long tile = (long long)blockIdx.x * 8 + wave;
    const int ti = (int)(tile >> 9);          // tile / 512
    const int tj = (int)(tile & (NT - 1));
    const int ibase = ti * 16, jbase = tj * 16;
    const int half = lane >> 4;               // 0 or 1
    const int l16  = lane & 15;

    // A fragment: lane holds row m = l16; elems 0..7 = K[half*16 .. +7],
    //             elems 8..15 = K[half*16+8 .. +7]
    // B fragment: lane holds column n = l16; 16 contiguous K at half*16.
    const unsigned short* arow_hi = h_hi + (size_t)(ibase + l16) * HID;
    const unsigned short* arow_lo = h_lo + (size_t)(ibase + l16) * HID;
    const unsigned short* brow_hi = h_hi + (size_t)(jbase + l16) * HID;
    const unsigned short* brow_lo = h_lo + (size_t)(jbase + l16) * HID;

    v8f acc = {};
    #pragma unroll
    for (int kt = 0; kt < HID; kt += 32) {
        const v8bf  a_hi0 = *reinterpret_cast<const v8bf*>(arow_hi + kt + half * 16);
        const v8bf  a_hi1 = *reinterpret_cast<const v8bf*>(arow_hi + kt + half * 16 + 8);
        const v8bf  a_lo0 = *reinterpret_cast<const v8bf*>(arow_lo + kt + half * 16);
        const v8bf  a_lo1 = *reinterpret_cast<const v8bf*>(arow_lo + kt + half * 16 + 8);
        const v16bf b_hi  = *reinterpret_cast<const v16bf*>(brow_hi + kt + half * 16);
        const v16bf b_lo  = *reinterpret_cast<const v16bf*>(brow_lo + kt + half * 16);

        v16bf a_h, a_l;
        #pragma unroll
        for (int i = 0; i < 8; ++i) {
            a_h[i] = a_hi0[i]; a_h[i + 8] = a_hi1[i];
            a_l[i] = a_lo0[i]; a_l[i + 8] = a_lo1[i];
        }
        // D = A*B + C  (f32 accumulate), 8-arg probe-confirmed signature
        acc = __builtin_amdgcn_wmma_f32_16x16x32_bf16(false, a_h, false, b_hi,
                                                      (short)0, acc, false, false);
        acc = __builtin_amdgcn_wmma_f32_16x16x32_bf16(false, a_h, false, b_lo,
                                                      (short)0, acc, false, false);
        acc = __builtin_amdgcn_wmma_f32_16x16x32_bf16(false, a_l, false, b_hi,
                                                      (short)0, acc, false, false);
    }

    const float thr = *thr_p;
    const int   J   = jbase + l16;            // C/D layout: column = lane&15
    const float sqJ = sq[J];

    float* probs    = out;                    // [0, M)
    float* iu       = out + M_PAIRS;          // [M, 2M)
    float* ju       = out + 2 * M_PAIRS;      // [2M, 3M)
    float* soft_adj = out + 3 * M_PAIRS;      // [3M, 3M + N^2)

    #pragma unroll
    for (int e = 0; e < 8; ++e) {             // C/D layout: row = e + 8*half
        const int I = ibase + e + half * 8;
        const float g  = acc[e];
        const float d2 = sq[I] + sqJ - 2.0f * g;
        float p = 0.0f, adj = 0.0f;
        if (I != J) {
            const float dist = sqrtf(fmaxf(d2, 1e-12f));
            p   = 1.0f / (1.0f + __expf(dist - thr));  // sigmoid(thr - dist)
            adj = p;
        }
        soft_adj[(size_t)I * N_NODES + J] = adj;
        if (I < J) {
            const long long t = (long long)I * N_NODES
                              - (long long)I * (I + 1) / 2 + (J - I - 1);
            probs[t] = p;
            iu[t]    = (float)I;
            ju[t]    = (float)J;
        }
    }
}

extern "C" void kernel_launch(void* const* d_in, const int* in_sizes, int n_in,
                              void* d_out, int out_size, void* d_ws, size_t ws_size,
                              hipStream_t stream) {
    const float* X   = (const float*)d_in[0];   // (8192, 64)
    const float* W   = (const float*)d_in[1];   // (64, 128)
    const float* b   = (const float*)d_in[2];   // (128,)
    const float* thr = (const float*)d_in[3];   // scalar
    float* out = (float*)d_out;

    // workspace: h_hi (2MB) | h_lo (2MB) | sq (32KB)
    unsigned short* h_hi = (unsigned short*)d_ws;
    unsigned short* h_lo = h_hi + (size_t)N_NODES * HID;
    float* sq = (float*)(h_lo + (size_t)N_NODES * HID);

    encode_split_kernel<<<N_NODES / 4, 128, 0, stream>>>(X, W, b, h_hi, h_lo, sq);

    dim3 blk(32, 8);                          // 8 wave32 per block
    const int n_tiles = NT * NT;              // 262144
    pair_wmma_kernel<<<n_tiles / 8, blk, 0, stream>>>(h_hi, h_lo, sq, thr, out);
}
